// CrossAttention_87668872446062
// MI455X (gfx1250) — matrix operand order repaired and checked by reading, not compiled, not run on hardware
//
#include <hip/hip_runtime.h>
#include <hip/hip_bf16.h>
#include <math.h>

// ---------------------------------------------------------------------------
// CDNA5 (gfx1250) fused epipolar cross-attention block.
// GEMMs use v_wmma_f32_16x16x32_bf16 (wave32, 16x64 per wave, f32 accum,
// K templated for full unroll / deep load-wmma pipelining).
// Geometry candidate compaction uses wave32 ballot/mbcnt.
// ---------------------------------------------------------------------------

typedef __bf16 bf16_t;
typedef __attribute__((ext_vector_type(16))) __bf16 v16bf;
typedef __attribute__((ext_vector_type(8)))  __bf16 v8bf;
typedef __attribute__((ext_vector_type(8)))  float   v8f;

#define DMODEL 256
#define NHEAD  8
#define HDIM   32
#define CANDMAX 240      // max(H,W)*AREA_WIDTH for H=W=48
#define LN_EPS 1e-5f

// ------------------------------ geometry -----------------------------------

__device__ inline void inv3(const float* m, float* o) {
  float a=m[0],b=m[1],c=m[2],d=m[3],e=m[4],f=m[5],g=m[6],h=m[7],i=m[8];
  float A =  e*i - f*h;
  float B = -(d*i - f*g);
  float Cc = d*h - e*g;
  float det = a*A + b*B + c*Cc;
  float id = 1.0f / det;
  o[0]=A*id;          o[1]=-(b*i-c*h)*id; o[2]=(b*f-c*e)*id;
  o[3]=B*id;          o[4]=(a*i-c*g)*id;  o[5]=-(a*f-c*d)*id;
  o[6]=Cc*id;         o[7]=-(a*h-b*g)*id; o[8]=(a*e-b*d)*id;
}

__device__ inline void mm3(const float* a, const float* b, float* o) {
  for (int i=0;i<3;++i)
    for (int j=0;j<3;++j)
      o[i*3+j] = a[i*3+0]*b[0*3+j] + a[i*3+1]*b[1*3+j] + a[i*3+2]*b[2*3+j];
}

__global__ void k_compute_F(const float* __restrict__ K0, const float* __restrict__ K1,
                            const float* __restrict__ R, const float* __restrict__ t,
                            const int* __restrict__ scale_p, float* __restrict__ F) {
  float sc = (float)(*scale_p);
  float A[9], B[9];
  for (int i=0;i<9;++i) { A[i]=K0[i]; B[i]=K1[i]; }
  for (int i=0;i<6;++i) { A[i]/=sc; B[i]/=sc; }   // first two rows / scale
  float Ai[9], Bi[9];
  inv3(A, Ai);
  inv3(B, Bi);
  float S[9] = { 0.f,  -t[2],  t[1],
                 t[2],  0.f,  -t[0],
                -t[1],  t[0],  0.f };
  float T1[9], T2[9], BiT[9];
  mm3(S, R, T1);          // skew(t) @ R
  mm3(T1, Ai, T2);        // @ inv(K0s)
  BiT[0]=Bi[0]; BiT[1]=Bi[3]; BiT[2]=Bi[6];
  BiT[3]=Bi[1]; BiT[4]=Bi[4]; BiT[5]=Bi[7];
  BiT[6]=Bi[2]; BiT[7]=Bi[5]; BiT[8]=Bi[8];
  float Fo[9];
  mm3(BiT, T2, Fo);       // inv(K1s)^T @ ...
  for (int i=0;i<9;++i) F[i] = Fo[i];
}

__global__ void k_lines(const float* __restrict__ F, const int* __restrict__ Wp,
                        int L, float* __restrict__ lines) {
  int s = blockIdx.x * blockDim.x + threadIdx.x;
  if (s >= L) return;
  int W = *Wp;
  float xs = (float)(s % W), ys = (float)(s / W);
  float l0 = F[0]*xs + F[1]*ys + F[2];
  float l1 = F[3]*xs + F[4]*ys + F[5];
  float l2 = F[6]*xs + F[7]*ys + F[8];
  float n = sqrtf(l0*l0 + l1*l1) + 1e-8f;
  float a = l0/n, b = l1/n, c = l2/n;
  lines[s*4+0] = a;
  lines[s*4+1] = b;
  lines[s*4+2] = c;
  lines[s*4+3] = (fabsf(b) >= fabsf(a)) ? 1.0f : 0.0f;
}

// Ordered stream compaction of in-band candidates: one wave32 per query.
__global__ void k_candidates(const float* __restrict__ lines, int* __restrict__ idx,
                             int* __restrict__ valid, const int* __restrict__ Wp,
                             const int* __restrict__ Hp, int C) {
  int qq = blockIdx.x;
  int W = *Wp, H = *Hp, S = W * H;
  float a = lines[qq*4+0], b = lines[qq*4+1], c = lines[qq*4+2];
  bool mode = lines[qq*4+3] > 0.5f;
  float bs = (fabsf(b) > 1e-8f) ? b : 1e-8f;
  float as = (fabsf(a) > 1e-8f) ? a : 1e-8f;
  int lane = threadIdx.x;
  int count = 0;
  for (int base = 0; base < S; base += 32) {
    int p = base + lane;
    bool within = false;
    if (p < S) {
      float xs = (float)(p % W), ys = (float)(p / W);
      float ypred = -(a*xs + c) / bs;
      float xpred = -(b*ys + c) / as;
      bool wy = (ys < ypred + 2.0f) && (ys > ypred - 2.0f);
      bool wx = (xs < xpred + 2.0f) && (xs > xpred - 2.0f);
      within = mode ? wy : wx;
    }
    unsigned mask = __builtin_amdgcn_ballot_w32(within);
    unsigned pre  = __builtin_amdgcn_mbcnt_lo(mask, 0u);
    if (within) {
      int pos = count + (int)pre;
      if (pos < C) {
        idx[(size_t)qq*C + pos]   = p;
        valid[(size_t)qq*C + pos] = 1;
      }
    }
    count += __popc(mask);
  }
}

// ------------------------------ conversions --------------------------------

__global__ void k_f32_to_bf16(const float* __restrict__ in, bf16_t* __restrict__ out, int n) {
  int i = blockIdx.x * blockDim.x + threadIdx.x;
  if (i < n) out[i] = (bf16_t)in[i];
}

// W is KxN row-major f32; out is NxK bf16 (so WMMA B fragment is contiguous)
__global__ void k_transpose_to_bf16(const float* __restrict__ in, bf16_t* __restrict__ out,
                                    int K, int N) {
  int i = blockIdx.x * blockDim.x + threadIdx.x;
  if (i < K * N) {
    int kk = i / N, nn = i % N;
    out[(size_t)nn * K + kk] = (bf16_t)in[i];
  }
}

__global__ void k_relu_to_bf16(const float* __restrict__ in, bf16_t* __restrict__ out, int n) {
  int i = blockIdx.x * blockDim.x + threadIdx.x;
  if (i < n) out[i] = (bf16_t)fmaxf(in[i], 0.0f);
}

// ------------------------------ WMMA GEMM ----------------------------------
// C[MxN] (f32) = A[MxK] (bf16, row-major) * Bt[NxK] (bf16, i.e. B transposed)
// Each wave computes a 16x64 strip: per 32-wide k-step the A fragment and all
// four B fragments are issued before the wmma group, and K is a compile-time
// constant so the fully unrolled loop lets the scheduler overlap the next
// k-step's loads with the current wmmas.
// Fragment lane maps per ISA 7.12.2 (16-bit A 16x32; B 32x16; f32 C 16x16).
template <int K>
__global__ void __launch_bounds__(256)
k_gemm_bf16(const bf16_t* __restrict__ A, const bf16_t* __restrict__ Bt,
            float* __restrict__ C, int M, int N) {
  int wave = threadIdx.x >> 5;
  int lane = threadIdx.x & 31;
  int tile = blockIdx.x * (blockDim.x >> 5) + wave;
  int groupsN = N >> 6;                      // 64-wide N groups
  int tm = tile / groupsN;
  int tg = tile % groupsN;
  if (tm >= (M >> 4)) return;

  int l16 = lane & 15;
  int kh  = lane >> 4;                       // K-half select (0/1)
  const bf16_t* arow = A + (size_t)((tm << 4) + l16) * K;
  const bf16_t* bcol[4];
#pragma unroll
  for (int j = 0; j < 4; ++j)
    bcol[j] = Bt + (size_t)((tg << 6) + (j << 4) + l16) * K;

  v8f acc[4] = {};
#pragma unroll
  for (int kb = 0; kb < K; kb += 32) {
    __builtin_prefetch(arow + kb + 128, 0, 2);
    // Issue A + all four B fragment loads before any wmma of this k-step.
    // A 16x32 bf16 fragment: lanes 0-15 -> K {kb..kb+7, kb+16..kb+23},
    //                        lanes16-31 -> K {kb+8..15, kb+24..31}
    v8bf a_lo = *reinterpret_cast<const v8bf*>(arow + kb + kh * 8);
    v8bf a_hi = *reinterpret_cast<const v8bf*>(arow + kb + 16 + kh * 8);
    v8bf b_lo[4], b_hi[4];
#pragma unroll
    for (int j = 0; j < 4; ++j) {
      // B 32x16 fragment: lanes 0-15 -> K {kb..kb+7, kb+16..kb+23}, lanes16-31 -> K {kb+8..15, kb+24..31}
      b_lo[j] = *reinterpret_cast<const v8bf*>(bcol[j] + kb + kh * 8);
      b_hi[j] = *reinterpret_cast<const v8bf*>(bcol[j] + kb + 16 + kh * 8);
    }
    v16bf af;
#pragma unroll
    for (int e = 0; e < 8; ++e) { af[e] = a_lo[e]; af[e + 8] = a_hi[e]; }
#pragma unroll
    for (int j = 0; j < 4; ++j) {
      v16bf bf;
#pragma unroll
      for (int e = 0; e < 8; ++e) { bf[e] = b_lo[j][e]; bf[e + 8] = b_hi[j][e]; }
      acc[j] = __builtin_amdgcn_wmma_f32_16x16x32_bf16(
          /*neg_a=*/false, af, /*neg_b=*/false, bf,
          /*c_mod=*/(short)0, acc[j], /*reuse_a=*/false, /*reuse_b=*/false);
    }
  }
  // f32 C/D layout: VGPR r -> row r (lanes 0-15) or 8+r (lanes 16-31)
  int crow = (tm << 4) + 8 * kh;
#pragma unroll
  for (int j = 0; j < 4; ++j) {
    int col = (tg << 6) + (j << 4) + l16;
#pragma unroll
    for (int r = 0; r < 8; ++r)
      C[(size_t)(crow + r) * N + col] = acc[j][r];
  }
}

// ------------------------------ attention ----------------------------------
// One 256-thread block per query: gather scores over C candidates (8 heads,
// channel layout d = j*8 + h), masked softmax, weighted sum of gathered v.
__global__ void __launch_bounds__(256)
k_attention(const float* __restrict__ q, const float* __restrict__ k,
            const float* __restrict__ v, const int* __restrict__ idx,
            const int* __restrict__ valid, float* __restrict__ msg, int C) {
  __shared__ float qs[DMODEL];
  __shared__ float sc[CANDMAX * NHEAD];
  __shared__ int   is[CANDMAX];
  __shared__ float red[2 * NHEAD];
  int l = blockIdx.x;
  int t = threadIdx.x;

  qs[t] = q[(size_t)l * DMODEL + t] * 0.17677669529663687f;   // 1/sqrt(32)
  if (t < C) is[t] = idx[(size_t)l * C + t];
  __syncthreads();

  if (t < C) {
    int i = is[t];
    bool vf = valid[(size_t)l * C + t] != 0;
    const float* krow = k + (size_t)i * DMODEL;
    float accs[NHEAD];
#pragma unroll
    for (int h = 0; h < NHEAD; ++h) accs[h] = 0.0f;
    for (int j = 0; j < HDIM; ++j) {
#pragma unroll
      for (int h = 0; h < NHEAD; ++h)
        accs[h] += qs[j * NHEAD + h] * krow[j * NHEAD + h];
    }
#pragma unroll
    for (int h = 0; h < NHEAD; ++h)
      sc[t * NHEAD + h] = vf ? accs[h] : -1e9f;
  }
  __syncthreads();

  if (t < NHEAD) {
    float m = -INFINITY;
    for (int cc = 0; cc < C; ++cc) m = fmaxf(m, sc[cc * NHEAD + t]);
    float s = 0.0f;
    for (int cc = 0; cc < C; ++cc) s += expf(sc[cc * NHEAD + t] - m);
    red[t] = m;
    red[NHEAD + t] = 1.0f / s;
  }
  __syncthreads();

  for (int e = t; e < C * NHEAD; e += blockDim.x) {
    int h = e & (NHEAD - 1);
    sc[e] = expf(sc[e] - red[h]) * red[NHEAD + h];
  }
  __syncthreads();

  int h = t & (NHEAD - 1);            // channel d = j*8 + h  ->  head = d % 8
  float acc = 0.0f;
  for (int cc = 0; cc < C; ++cc)
    acc += sc[cc * NHEAD + h] * v[(size_t)is[cc] * DMODEL + t];
  msg[(size_t)l * DMODEL + t] = acc;
}

// ------------------------------ layernorm ----------------------------------

__device__ inline float block_sum256(float vv, float* buf) {
  int t = threadIdx.x;
  buf[t] = vv;
  __syncthreads();
  for (int s = 128; s > 0; s >>= 1) {
    if (t < s) buf[t] += buf[t + s];
    __syncthreads();
  }
  float r = buf[0];
  __syncthreads();
  return r;
}

// LN over mergeOut rows, then build bf16 concat [x | ln(mergeOut)] for MLP1.
__global__ void __launch_bounds__(256)
k_ln_cat(const float* __restrict__ mo, const float* __restrict__ g,
         const float* __restrict__ b, const float* __restrict__ x,
         bf16_t* __restrict__ catb) {
  __shared__ float buf[256];
  int l = blockIdx.x, t = threadIdx.x;
  float vv = mo[(size_t)l * DMODEL + t];
  float mean = block_sum256(vv, buf) * (1.0f / DMODEL);
  float dd = vv - mean;
  float var = block_sum256(dd * dd, buf) * (1.0f / DMODEL);
  float y = dd * rsqrtf(var + LN_EPS) * g[t] + b[t];
  catb[(size_t)l * (2 * DMODEL) + t]          = (bf16_t)x[(size_t)l * DMODEL + t];
  catb[(size_t)l * (2 * DMODEL) + DMODEL + t] = (bf16_t)y;
}

// LN over final MLP output rows, add residual x, write f32 out.
__global__ void __launch_bounds__(256)
k_ln_residual(const float* __restrict__ o2, const float* __restrict__ g,
              const float* __restrict__ b, const float* __restrict__ x,
              float* __restrict__ out) {
  __shared__ float buf[256];
  int l = blockIdx.x, t = threadIdx.x;
  float vv = o2[(size_t)l * DMODEL + t];
  float mean = block_sum256(vv, buf) * (1.0f / DMODEL);
  float dd = vv - mean;
  float var = block_sum256(dd * dd, buf) * (1.0f / DMODEL);
  float y = dd * rsqrtf(var + LN_EPS) * g[t] + b[t];
  out[(size_t)l * DMODEL + t] = x[(size_t)l * DMODEL + t] + y;
}

// ------------------------------ launcher -----------------------------------

extern "C" void kernel_launch(void* const* d_in, const int* in_sizes, int n_in,
                              void* d_out, int out_size, void* d_ws, size_t ws_size,
                              hipStream_t stream) {
  const float* x      = (const float*)d_in[0];
  const float* source = (const float*)d_in[1];
  const float* K0     = (const float*)d_in[2];
  const float* K1     = (const float*)d_in[3];
  const float* R      = (const float*)d_in[4];
  const float* t      = (const float*)d_in[5];
  const float* qW     = (const float*)d_in[6];
  const float* kW     = (const float*)d_in[7];
  const float* vW     = (const float*)d_in[8];
  const float* mW     = (const float*)d_in[9];
  const float* W1     = (const float*)d_in[10];
  const float* W2     = (const float*)d_in[11];
  const float* ln1g   = (const float*)d_in[12];
  const float* ln1b   = (const float*)d_in[13];
  const float* ln2g   = (const float*)d_in[14];
  const float* ln2b   = (const float*)d_in[15];
  const int*   Hp     = (const int*)d_in[16];
  const int*   Wp     = (const int*)d_in[17];
  const int*   scp    = (const int*)d_in[18];

  const int L = in_sizes[0] / DMODEL;   // N*S (N=1, S=H*W=2304)
  const int C = CANDMAX;                // max(H,W)*AREA_WIDTH = 240

  // --- workspace carve (256B aligned) ---
  char* p = (char*)d_ws;
  auto carve = [&](size_t bytes) -> void* {
    void* r = (void*)p;
    p += (bytes + 255) & ~(size_t)255;
    return r;
  };
  float*  Fm     = (float*)carve(16 * sizeof(float));
  float*  lines  = (float*)carve((size_t)L * 4 * sizeof(float));
  int*    idx    = (int*)carve((size_t)L * C * sizeof(int));
  int*    valid  = (int*)carve((size_t)L * C * sizeof(int));
  bf16_t* xb     = (bf16_t*)carve((size_t)L * DMODEL * sizeof(bf16_t));
  bf16_t* sb     = (bf16_t*)carve((size_t)L * DMODEL * sizeof(bf16_t));
  bf16_t* WqT    = (bf16_t*)carve((size_t)DMODEL * DMODEL * sizeof(bf16_t));
  bf16_t* WkT    = (bf16_t*)carve((size_t)DMODEL * DMODEL * sizeof(bf16_t));
  bf16_t* WvT    = (bf16_t*)carve((size_t)DMODEL * DMODEL * sizeof(bf16_t));
  bf16_t* WmT    = (bf16_t*)carve((size_t)DMODEL * DMODEL * sizeof(bf16_t));
  bf16_t* W1T    = (bf16_t*)carve((size_t)(2*DMODEL) * (2*DMODEL) * sizeof(bf16_t));
  bf16_t* W2T    = (bf16_t*)carve((size_t)DMODEL * (2*DMODEL) * sizeof(bf16_t));
  float*  qf     = (float*)carve((size_t)L * DMODEL * sizeof(float));
  float*  kf     = (float*)carve((size_t)L * DMODEL * sizeof(float));
  float*  vf     = (float*)carve((size_t)L * DMODEL * sizeof(float));
  float*  msg    = (float*)carve((size_t)L * DMODEL * sizeof(float));
  bf16_t* msgb   = (bf16_t*)carve((size_t)L * DMODEL * sizeof(bf16_t));
  float*  mo     = (float*)carve((size_t)L * DMODEL * sizeof(float));
  bf16_t* catb   = (bf16_t*)carve((size_t)L * 2 * DMODEL * sizeof(bf16_t));
  float*  hbuf   = (float*)carve((size_t)L * 2 * DMODEL * sizeof(float));
  bf16_t* hb     = (bf16_t*)carve((size_t)L * 2 * DMODEL * sizeof(bf16_t));
  float*  o2     = (float*)carve((size_t)L * DMODEL * sizeof(float));
  (void)ws_size; (void)n_in;

  // --- geometry / candidate selection ---
  hipMemsetAsync(idx,   0, (size_t)L * C * sizeof(int), stream);
  hipMemsetAsync(valid, 0, (size_t)L * C * sizeof(int), stream);
  k_compute_F<<<1, 1, 0, stream>>>(K0, K1, R, t, scp, Fm);
  k_lines<<<(L + 255) / 256, 256, 0, stream>>>(Fm, Wp, L, lines);
  k_candidates<<<L, 32, 0, stream>>>(lines, idx, valid, Wp, Hp, C);

  // --- bf16 conversions ---
  int nA = L * DMODEL;
  k_f32_to_bf16<<<(nA + 255) / 256, 256, 0, stream>>>(x, xb, nA);
  k_f32_to_bf16<<<(nA + 255) / 256, 256, 0, stream>>>(source, sb, nA);
  int nW = DMODEL * DMODEL;
  k_transpose_to_bf16<<<(nW + 255) / 256, 256, 0, stream>>>(qW, WqT, DMODEL, DMODEL);
  k_transpose_to_bf16<<<(nW + 255) / 256, 256, 0, stream>>>(kW, WkT, DMODEL, DMODEL);
  k_transpose_to_bf16<<<(nW + 255) / 256, 256, 0, stream>>>(vW, WvT, DMODEL, DMODEL);
  k_transpose_to_bf16<<<(nW + 255) / 256, 256, 0, stream>>>(mW, WmT, DMODEL, DMODEL);
  int nW1 = (2*DMODEL) * (2*DMODEL);
  k_transpose_to_bf16<<<(nW1 + 255) / 256, 256, 0, stream>>>(W1, W1T, 2*DMODEL, 2*DMODEL);
  int nW2 = (2*DMODEL) * DMODEL;
  k_transpose_to_bf16<<<(nW2 + 255) / 256, 256, 0, stream>>>(W2, W2T, 2*DMODEL, DMODEL);

  auto gemm256 = [&](const bf16_t* A, const bf16_t* Bt, float* Cm, int M, int N) {
    int tiles = (M / 16) * (N / 64);    // one wave per 16x64 strip
    int blocks = (tiles + 7) / 8;       // 8 waves per 256-thread block
    k_gemm_bf16<256><<<blocks, 256, 0, stream>>>(A, Bt, Cm, M, N);
  };
  auto gemm512 = [&](const bf16_t* A, const bf16_t* Bt, float* Cm, int M, int N) {
    int tiles = (M / 16) * (N / 64);
    int blocks = (tiles + 7) / 8;
    k_gemm_bf16<512><<<blocks, 256, 0, stream>>>(A, Bt, Cm, M, N);
  };

  // --- q/k/v projections ---
  gemm256(xb, WqT, qf, L, DMODEL);
  gemm256(sb, WkT, kf, L, DMODEL);
  gemm256(sb, WvT, vf, L, DMODEL);

  // --- gather attention ---
  k_attention<<<L, 256, 0, stream>>>(qf, kf, vf, idx, valid, msg, C);

  // --- merge + LN1 + concat ---
  k_f32_to_bf16<<<(nA + 255) / 256, 256, 0, stream>>>(msg, msgb, nA);
  gemm256(msgb, WmT, mo, L, DMODEL);
  k_ln_cat<<<L, 256, 0, stream>>>(mo, ln1g, ln1b, x, catb);

  // --- MLP ---
  gemm512(catb, W1T, hbuf, L, 2*DMODEL);
  int nH = L * 2 * DMODEL;
  k_relu_to_bf16<<<(nH + 255) / 256, 256, 0, stream>>>(hbuf, hb, nH);
  gemm512(hb, W2T, o2, L, DMODEL);

  // --- LN2 + residual ---
  k_ln_residual<<<L, 256, 0, stream>>>(o2, ln2g, ln2b, x, (float*)d_out);
}
